// MGGATRecommender_69544110457225
// MI455X (gfx1250) — compile-verified, matrix operand order checked by reading, not run on hardware
//
#include <hip/hip_runtime.h>
#include <hip/hip_bf16.h>

typedef __bf16 bf16_t;
typedef bf16_t v16bf __attribute__((ext_vector_type(16)));
typedef bf16_t v8bf  __attribute__((ext_vector_type(8)));
typedef bf16_t v4bf  __attribute__((ext_vector_type(4)));
typedef float  v8f   __attribute__((ext_vector_type(8)));

static __device__ __forceinline__ v16bf ldfrag(const bf16_t* __restrict__ p) {
    v8bf lo = *(const v8bf*)p;          // K-octet {kb..kb+7}
    v8bf hi = *(const v8bf*)(p + 16);   // K-octet {kb+16..kb+23}
    return __builtin_shufflevector(lo, hi, 0,1,2,3,4,5,6,7,8,9,10,11,12,13,14,15);
}

// ---------------------------------------------------------------------------
// bf16 WMMA GEMM, K fixed = 128. A row-major bf16 [M,128]; Bt column-major
// bf16 [N,128] (Bt[n*128+k] = B[k,n]).  One wave computes 64x16 of C
// (4 M-subtiles sharing the B fragment in registers -> 4x B reuse).
// C[M,N] = act(A@B (+ A2@B2) (+ bias)) (+ resid).  f32 accumulate.
// ---------------------------------------------------------------------------
__global__ void __launch_bounds__(32)
wmma_gemm_bf16_kernel(const bf16_t* __restrict__ A,  const bf16_t* __restrict__ Bt,
                      const bf16_t* __restrict__ A2, const bf16_t* __restrict__ B2t,
                      const float* __restrict__ bias, const float* __restrict__ resid,
                      float* __restrict__ C, int M, int N, int act) {
    const int ln   = threadIdx.x;
    const int trow = blockIdx.y * 64;
    const int tcol = blockIdx.x * 16;
    const int r    = ln & 15;
    const int kb   = (ln >> 4) * 8;

    const bf16_t* Bcol = Bt + (size_t)(tcol + r) * 128 + kb;
    size_t aoff[4];
    #pragma unroll
    for (int t = 0; t < 4; ++t) {
        int a = trow + t * 16 + r;
        if (a >= M) a = M - 1;                       // clamp, guarded store
        aoff[t] = (size_t)a * 128 + kb;
    }

    v8f acc[4] = {};
    #pragma unroll
    for (int k0 = 0; k0 < 128; k0 += 32) {
        v16bf b = ldfrag(Bcol + k0);
        #pragma unroll
        for (int t = 0; t < 4; ++t) {
            v16bf a = ldfrag(A + aoff[t] + k0);
            acc[t] = __builtin_amdgcn_wmma_f32_16x16x32_bf16(false, a, false, b,
                                                             (short)0, acc[t], false, false);
        }
    }
    if (A2) {
        const bf16_t* B2col = B2t + (size_t)(tcol + r) * 128 + kb;
        #pragma unroll
        for (int k0 = 0; k0 < 128; k0 += 32) {
            v16bf b = ldfrag(B2col + k0);
            #pragma unroll
            for (int t = 0; t < 4; ++t) {
                v16bf a = ldfrag(A2 + aoff[t] + k0);
                acc[t] = __builtin_amdgcn_wmma_f32_16x16x32_bf16(false, a, false, b,
                                                                 (short)0, acc[t], false, false);
            }
        }
    }

    const int ccol = tcol + (ln & 15);
    const int rhalf = (ln >> 4) * 8;
    #pragma unroll
    for (int t = 0; t < 4; ++t) {
        const int crow0 = trow + t * 16 + rhalf;
        #pragma unroll
        for (int rr = 0; rr < 8; ++rr) {
            int crow = crow0 + rr;
            if (crow < M) {
                float v = acc[t][rr];
                if (bias)  v += bias[ccol];
                if (act)   v = v > 0.f ? v : (__expf(v) - 1.f);     // ELU
                if (resid) v += resid[(size_t)crow * N + ccol];
                C[(size_t)crow * N + ccol] = v;
            }
        }
    }
}

// streaming f32 -> bf16 (hardware cvt), 4 elements / thread
__global__ void __launch_bounds__(256)
cvt_bf16_kernel(const float* __restrict__ src, bf16_t* __restrict__ dst, long long n4) {
    long long i = (long long)blockIdx.x * 256 + threadIdx.x;
    if (i >= n4) return;
    float4 f = ((const float4*)src)[i];
    v4bf o;
    o[0] = (bf16_t)f.x; o[1] = (bf16_t)f.y; o[2] = (bf16_t)f.z; o[3] = (bf16_t)f.w;
    ((v4bf*)dst)[i] = o;
}

// weights: f32 [K,N] row-major -> bf16 [N,K] column-major-of-B
__global__ void __launch_bounds__(256)
transpose_cvt_kernel(const float* __restrict__ src, bf16_t* __restrict__ dst, int K, int N) {
    int idx = blockIdx.x * 256 + threadIdx.x;
    if (idx >= K * N) return;
    int k = idx / N, n = idx - k * N;
    dst[(size_t)n * K + k] = (bf16_t)src[idx];
}

// per-node dots: ds[n] = H1[n,:]·a_s ; dn[n] = H1[n,:]·a_n  (one wave / node)
__global__ void __launch_bounds__(256)
node_dots_kernel(const float* __restrict__ H1, const float* __restrict__ a_s,
                 const float* __restrict__ a_n, float* __restrict__ ds,
                 float* __restrict__ dn, int Nn) {
    int wave = blockIdx.x * 8 + (threadIdx.x >> 5);
    int ln   = threadIdx.x & 31;
    if (wave >= Nn) return;
    const float* row = H1 + (size_t)wave * 128;
    float ps = 0.f, pn = 0.f;
    #pragma unroll
    for (int j = 0; j < 4; ++j) {
        int i = ln + 32 * j;
        float h = row[i];
        ps += h * a_s[i];
        pn += h * a_n[i];
    }
    #pragma unroll
    for (int off = 16; off > 0; off >>= 1) {
        ps += __shfl_down(ps, off, 32);
        pn += __shfl_down(pn, off, 32);
    }
    if (ln == 0) { ds[wave] = ps; dn[wave] = pn; }
}

__global__ void __launch_bounds__(256)
fill_f32_kernel(float* __restrict__ p, float v, long long n) {
    long long i = (long long)blockIdx.x * 256 + threadIdx.x;
    if (i < n) p[i] = v;
}

// sc = leaky(ω·(ds[dst]+dn[src])) ; atomicMax of monotone-encoded sc per dst
__global__ void __launch_bounds__(256)
edge_score_kernel(const int* __restrict__ src, const int* __restrict__ dst,
                  const float* __restrict__ ds, const float* __restrict__ dn,
                  const float* __restrict__ counts, const float* __restrict__ omega,
                  int G, float* __restrict__ sc_out, unsigned* __restrict__ mx, int E) {
    int e = blockIdx.x * 256 + threadIdx.x;
    if (e >= E) return;
    int d = dst[e];
    float s = ds[d] + dn[src[e]];
    if (counts) {
        float w = 0.f;
        for (int g = 0; g < G; ++g) w += counts[(size_t)e * G + g] * omega[g];
        s *= w;
    }
    s = s > 0.f ? s : 0.2f * s;                    // leaky_relu(0.2)
    sc_out[e] = s;
    unsigned bits = __float_as_uint(s);
    unsigned enc  = (bits & 0x80000000u) ? ~bits : (bits | 0x80000000u);
    atomicMax(&mx[d], enc);
}

// w = exp(sc - max[dst]) in place; atomicAdd into segment sum
__global__ void __launch_bounds__(256)
edge_exp_kernel(const int* __restrict__ dst, float* __restrict__ sc,
                const unsigned* __restrict__ mx, float* __restrict__ ssum, int E) {
    int e = blockIdx.x * 256 + threadIdx.x;
    if (e >= E) return;
    int d = dst[e];
    unsigned u = mx[d];
    unsigned bits = (u & 0x80000000u) ? (u & 0x7FFFFFFFu) : ~u;
    float w = __expf(sc[e] - __uint_as_float(bits));
    sc[e] = w;
    atomicAdd(&ssum[d], w);
}

// one wave per edge: H2[dst] += alpha * H1[src]   (coalesced 128-float rows)
__global__ void __launch_bounds__(256)
edge_agg_kernel(const int* __restrict__ src, const int* __restrict__ dst,
                const float* __restrict__ w, const float* __restrict__ ssum,
                const float* __restrict__ H1, float* __restrict__ H2, int E) {
    long long idx = (long long)blockIdx.x * 256 + threadIdx.x;
    int e  = (int)(idx >> 5);
    int ln = (int)(idx & 31);
    if (e >= E) return;
    int s = src[e], d = dst[e];
    float alpha = w[e] / ssum[d];
    const float* hrow = H1 + (size_t)s * 128;
    float*       orow = H2 + (size_t)d * 128;
    #pragma unroll
    for (int j = 0; j < 4; ++j) {
        int f = ln + 32 * j;
        atomicAdd(&orow[f], hrow[f] * alpha);
    }
}

// out[b] = 4*sigmoid(U[ui]·Bm[ii] + bias_u[ui] + bias_b[ii] + b_x) + 1
__global__ void __launch_bounds__(256)
predict_kernel(const int* __restrict__ ui, const int* __restrict__ ii,
               const float* __restrict__ U, const float* __restrict__ Bm,
               const float* __restrict__ bias_u, const float* __restrict__ bias_b,
               const float* __restrict__ bx, float* __restrict__ out, int Bn, int FD) {
    int b = blockIdx.x * 256 + threadIdx.x;
    if (b >= Bn) return;
    int u = ui[b], it = ii[b];
    const float* up = U  + (size_t)u  * FD;
    const float* bp = Bm + (size_t)it * FD;
    float dot = 0.f;
    for (int i = 0; i < FD; ++i) dot += up[i] * bp[i];
    float raw = dot + bias_u[u] + bias_b[it] + bx[0];
    out[b] = 4.f / (1.f + __expf(-raw)) + 1.f;
}

extern "C" void kernel_launch(void* const* d_in, const int* in_sizes, int n_in,
                              void* d_out, int out_size, void* d_ws, size_t ws_size,
                              hipStream_t stream) {
    const int*   ui      = (const int*)  d_in[0];
    const int*   ii      = (const int*)  d_in[1];
    const float* S_u     = (const float*)d_in[2];
    const float* S_b     = (const float*)d_in[3];
    const int*   eu      = (const int*)  d_in[4];
    const int*   eb      = (const int*)  d_in[5];
    const float* counts  = (const float*)d_in[6];
    const float* W1_u    = (const float*)d_in[7];
    const float* a_s_u   = (const float*)d_in[8];
    const float* a_n_u   = (const float*)d_in[9];
    const float* W1_b    = (const float*)d_in[10];
    const float* a_s_b   = (const float*)d_in[11];
    const float* a_n_b   = (const float*)d_in[12];
    const float* omega   = (const float*)d_in[13];
    const float* W_u2    = (const float*)d_in[14];
    const float* W_us2_w = (const float*)d_in[15];
    const float* W_us2_b = (const float*)d_in[16];
    const float* W_b2    = (const float*)d_in[17];
    const float* W_bs2_w = (const float*)d_in[18];
    const float* W_bs2_b = (const float*)d_in[19];
    const float* W_u3    = (const float*)d_in[20];
    const float* W_b3    = (const float*)d_in[21];
    const float* H_u4    = (const float*)d_in[22];
    const float* H_b4    = (const float*)d_in[23];
    const float* bias_u  = (const float*)d_in[24];
    const float* bias_b  = (const float*)d_in[25];
    const float* b_x     = (const float*)d_in[26];

    const int Bn = in_sizes[0];
    const int NU = in_sizes[2] / 128;
    const int NI = in_sizes[3] / 128;
    const int EU = in_sizes[4] / 2;
    const int EB = in_sizes[5] / 2;
    const int G  = in_sizes[13];
    const int FD = in_sizes[20] / 128;

    const int* src_u = eu;           const int* dst_u = eu + EU;
    const int* src_b = eb;           const int* dst_b = eb + EB;

    // ---- workspace bump allocator (256B aligned) ----
    char*  ws  = (char*)d_ws;
    size_t off = 0;
    auto alloc = [&](size_t bytes) -> void* {
        void* p = ws + off;
        off += (bytes + 255) & ~(size_t)255;
        return p;
    };
    float*    H1u   = (float*)   alloc((size_t)NU * 128 * 4);  // H1 -> later H3 (f32)
    float*    H2u   = (float*)   alloc((size_t)NU * 128 * 4);  // H2 -> later U
    float*    H1b   = (float*)   alloc((size_t)NI * 128 * 4);
    float*    H2b   = (float*)   alloc((size_t)NI * 128 * 4);
    bf16_t*   Sbf_u = (bf16_t*)  alloc((size_t)NU * 128 * 2);  // bf16 activations
    bf16_t*   Sbf_b = (bf16_t*)  alloc((size_t)NI * 128 * 2);
    bf16_t*   Xbf_u = (bf16_t*)  alloc((size_t)NU * 128 * 2);  // bf16 of H2, then H3
    bf16_t*   Xbf_b = (bf16_t*)  alloc((size_t)NI * 128 * 2);
    float*    dsu   = (float*)   alloc((size_t)NU * 4);
    float*    dnu   = (float*)   alloc((size_t)NU * 4);
    unsigned* msu   = (unsigned*)alloc((size_t)NU * 4);
    float*    ssu   = (float*)   alloc((size_t)NU * 4);
    float*    dsb   = (float*)   alloc((size_t)NI * 4);
    float*    dnb   = (float*)   alloc((size_t)NI * 4);
    unsigned* msb   = (unsigned*)alloc((size_t)NI * 4);
    float*    ssb   = (float*)   alloc((size_t)NI * 4);
    float*    scu   = (float*)   alloc((size_t)EU * 4);        // scores -> exp weights
    float*    scb   = (float*)   alloc((size_t)EB * 4);
    bf16_t*   W1t_u = (bf16_t*)  alloc(128 * 128 * 2);         // transposed bf16 weights
    bf16_t*   W1t_b = (bf16_t*)  alloc(128 * 128 * 2);
    bf16_t*   W2t_u = (bf16_t*)  alloc(128 * 128 * 2);
    bf16_t*   Wst_u = (bf16_t*)  alloc(128 * 128 * 2);
    bf16_t*   W2t_b = (bf16_t*)  alloc(128 * 128 * 2);
    bf16_t*   Wst_b = (bf16_t*)  alloc(128 * 128 * 2);
    bf16_t*   W3t_u = (bf16_t*)  alloc(128 * (size_t)FD * 2);
    bf16_t*   W3t_b = (bf16_t*)  alloc(128 * (size_t)FD * 2);

    auto cdiv = [](long long a, long long b) { return (unsigned)((a + b - 1) / b); };

    // ---- zero / init accumulators ----
    fill_f32_kernel<<<cdiv((long long)NU * 128, 256), 256, 0, stream>>>(H2u, 0.f, (long long)NU * 128);
    fill_f32_kernel<<<cdiv((long long)NI * 128, 256), 256, 0, stream>>>(H2b, 0.f, (long long)NI * 128);
    fill_f32_kernel<<<cdiv(NU, 256), 256, 0, stream>>>(ssu, 0.f, NU);
    fill_f32_kernel<<<cdiv(NI, 256), 256, 0, stream>>>(ssb, 0.f, NI);
    fill_f32_kernel<<<cdiv(NU, 256), 256, 0, stream>>>((float*)msu, 0.f, NU);  // enc-min
    fill_f32_kernel<<<cdiv(NI, 256), 256, 0, stream>>>((float*)msb, 0.f, NI);

    // ---- bf16 conversions: activations + transposed weights ----
    cvt_bf16_kernel<<<cdiv((long long)NU * 32, 256), 256, 0, stream>>>(S_u, Sbf_u, (long long)NU * 32);
    cvt_bf16_kernel<<<cdiv((long long)NI * 32, 256), 256, 0, stream>>>(S_b, Sbf_b, (long long)NI * 32);
    transpose_cvt_kernel<<<64, 256, 0, stream>>>(W1_u,    W1t_u, 128, 128);
    transpose_cvt_kernel<<<64, 256, 0, stream>>>(W1_b,    W1t_b, 128, 128);
    transpose_cvt_kernel<<<64, 256, 0, stream>>>(W_u2,    W2t_u, 128, 128);
    transpose_cvt_kernel<<<64, 256, 0, stream>>>(W_us2_w, Wst_u, 128, 128);
    transpose_cvt_kernel<<<64, 256, 0, stream>>>(W_b2,    W2t_b, 128, 128);
    transpose_cvt_kernel<<<64, 256, 0, stream>>>(W_bs2_w, Wst_b, 128, 128);
    transpose_cvt_kernel<<<cdiv(128 * FD, 256), 256, 0, stream>>>(W_u3, W3t_u, 128, FD);
    transpose_cvt_kernel<<<cdiv(128 * FD, 256), 256, 0, stream>>>(W_b3, W3t_b, 128, FD);

    // ---- H1 = S @ W1 (bf16 WMMA, f32 accum) ----
    wmma_gemm_bf16_kernel<<<dim3(8, cdiv(NU, 64)), 32, 0, stream>>>(
        Sbf_u, W1t_u, nullptr, nullptr, nullptr, nullptr, H1u, NU, 128, 0);
    wmma_gemm_bf16_kernel<<<dim3(8, cdiv(NI, 64)), 32, 0, stream>>>(
        Sbf_b, W1t_b, nullptr, nullptr, nullptr, nullptr, H1b, NI, 128, 0);

    // ---- per-node attention dots ----
    node_dots_kernel<<<cdiv(NU, 8), 256, 0, stream>>>(H1u, a_s_u, a_n_u, dsu, dnu, NU);
    node_dots_kernel<<<cdiv(NI, 8), 256, 0, stream>>>(H1b, a_s_b, a_n_b, dsb, dnb, NI);

    // ---- edge scores + segment max ----
    edge_score_kernel<<<cdiv(EU, 256), 256, 0, stream>>>(
        src_u, dst_u, dsu, dnu, nullptr, nullptr, G, scu, msu, EU);
    edge_score_kernel<<<cdiv(EB, 256), 256, 0, stream>>>(
        src_b, dst_b, dsb, dnb, counts, omega, G, scb, msb, EB);

    // ---- exp + segment sum ----
    edge_exp_kernel<<<cdiv(EU, 256), 256, 0, stream>>>(dst_u, scu, msu, ssu, EU);
    edge_exp_kernel<<<cdiv(EB, 256), 256, 0, stream>>>(dst_b, scb, msb, ssb, EB);

    // ---- weighted aggregation: H2[dst] += alpha * H1[src] ----
    edge_agg_kernel<<<cdiv((long long)EU * 32, 256), 256, 0, stream>>>(
        src_u, dst_u, scu, ssu, H1u, H2u, EU);
    edge_agg_kernel<<<cdiv((long long)EB * 32, 256), 256, 0, stream>>>(
        src_b, dst_b, scb, ssb, H1b, H2b, EB);

    // ---- convert H2 -> bf16 ----
    cvt_bf16_kernel<<<cdiv((long long)NU * 32, 256), 256, 0, stream>>>(H2u, Xbf_u, (long long)NU * 32);
    cvt_bf16_kernel<<<cdiv((long long)NI * 32, 256), 256, 0, stream>>>(H2b, Xbf_b, (long long)NI * 32);

    // ---- H3 = elu(H2@W2 + S@Ws2 + b)  (fused double-K WMMA; into H1 buffer) ----
    wmma_gemm_bf16_kernel<<<dim3(8, cdiv(NU, 64)), 32, 0, stream>>>(
        Xbf_u, W2t_u, Sbf_u, Wst_u, W_us2_b, nullptr, H1u, NU, 128, 1);
    wmma_gemm_bf16_kernel<<<dim3(8, cdiv(NI, 64)), 32, 0, stream>>>(
        Xbf_b, W2t_b, Sbf_b, Wst_b, W_bs2_b, nullptr, H1b, NI, 128, 1);

    // ---- convert H3 -> bf16 (reuse X buffers) ----
    cvt_bf16_kernel<<<cdiv((long long)NU * 32, 256), 256, 0, stream>>>(H1u, Xbf_u, (long long)NU * 32);
    cvt_bf16_kernel<<<cdiv((long long)NI * 32, 256), 256, 0, stream>>>(H1b, Xbf_b, (long long)NI * 32);

    // ---- U = elu(H3@W3) + H4   (N = FD = 64; into H2 buffer) ----
    wmma_gemm_bf16_kernel<<<dim3(FD / 16, cdiv(NU, 64)), 32, 0, stream>>>(
        Xbf_u, W3t_u, nullptr, nullptr, nullptr, H_u4, H2u, NU, FD, 1);
    wmma_gemm_bf16_kernel<<<dim3(FD / 16, cdiv(NI, 64)), 32, 0, stream>>>(
        Xbf_b, W3t_b, nullptr, nullptr, nullptr, H_b4, H2b, NI, FD, 1);

    // ---- batched dot + sigmoid ----
    predict_kernel<<<cdiv(Bn, 256), 256, 0, stream>>>(
        ui, ii, H2u, H2b, bias_u, bias_b, b_x, (float*)d_out, Bn, FD);
}
